// GRUDecoder_2_65343632441416
// MI455X (gfx1250) — compile-verified, matrix operand order checked by reading, not run on hardware
//
#include <hip/hip_runtime.h>
#include <hip/hip_bf16.h>

// ---------------------------------------------------------------------------
// GRU decoder (NRI-style) for MI455X / gfx1250.
// All heavy matmuls run on v_wmma_f32_16x16x32_f16 (f16 in, f32 accumulate).
// Edge-message MLP is fused: pre -> fc1 -> relu -> fc2 -> relu -> *rel_type
// summed over K in registers, then scatter-added into aug via f32 atomics.
// k_msg uses 4 waves/block, 4 column-tiles each, to stay spill-free.
// Workspace use: ~76 MB.
// ---------------------------------------------------------------------------

typedef _Float16 h16;
typedef __attribute__((ext_vector_type(16))) _Float16 v16h;
typedef __attribute__((ext_vector_type(8)))  float    v8f;

#define Bb   8
#define Nn   32
#define Tt   50
#define Dd   4
#define Kk   2
#define MHc  256
#define MOc  256
#define NHc  256
#define H2c  260     // MO + D
#define H2p  288     // H2 padded to multiple of 32 for WMMA K
#define G3   780     // 3*H2
#define G3p  784     // padded to multiple of 16 (N tiles)
#define Ee   992
#define ETILES 62    // 992/16
#define BN_EPS 1e-5f

// ---------------- WMMA fragment helpers (per ISA 7.12.2 layouts) -----------
__device__ __forceinline__ int kbase_of(int g, int half) {
  return ((g < 4) ? (2 * g) : (16 + 2 * (g - 4))) + 8 * half;
}

// Load a 32-wide K slice for this lane's row (A: row=M; B: row=output col of W^T).
__device__ __forceinline__ v16h frag_ld(const h16* __restrict__ rp, int half) {
  v16h a;
#pragma unroll
  for (int g = 0; g < 8; ++g) {
    int kb = kbase_of(g, half);
    a[2 * g]     = rp[kb];
    a[2 * g + 1] = rp[kb + 1];
  }
  return a;
}

// Same, but zero beyond kmax (for pre, which is only K=8 wide).
__device__ __forceinline__ v16h frag_ld_kmax(const h16* __restrict__ rp, int half, int kmax) {
  v16h a;
#pragma unroll
  for (int g = 0; g < 8; ++g) {
    int kb = kbase_of(g, half);
    a[2 * g]     = (kb     < kmax) ? rp[kb]     : (h16)0.f;
    a[2 * g + 1] = (kb + 1 < kmax) ? rp[kb + 1] : (h16)0.f;
  }
  return a;
}

__device__ __forceinline__ v8f wmma16(v16h a, v16h b, v8f c) {
  return __builtin_amdgcn_wmma_f32_16x16x32_f16(false, a, false, b, (short)0, c, false, false);
}

__device__ __forceinline__ v8f zero8() {
  v8f c = {0.f, 0.f, 0.f, 0.f, 0.f, 0.f, 0.f, 0.f};
  return c;
}

__device__ __forceinline__ float sigf(float x) { return 1.f / (1.f + __expf(-x)); }
__device__ __forceinline__ float eluf(float x) { return x > 0.f ? x : (__expf(x) - 1.f); }

// ---------------- k_indices: one-hot -> edge endpoint indices --------------
__global__ void k_indices(const float* __restrict__ rel_send,
                          const float* __restrict__ rel_rec,
                          int* __restrict__ send_idx, int* __restrict__ recv_idx) {
  int e = blockIdx.x * blockDim.x + threadIdx.x;
  if (e >= Ee) return;
  int s = 0, r = 0;
  for (int n = 0; n < Nn; ++n) {
    if (rel_send[e * Nn + n] > 0.5f) s = n;
    if (rel_rec[e * Nn + n]  > 0.5f) r = n;
  }
  send_idx[e] = s;
  recv_idx[e] = r;
}

// ---------------- k_pack: f32 weights -> zero-padded f16 copies ------------
__global__ void k_pack(const float* __restrict__ fc1w, const float* __restrict__ fc2w,
                       const float* __restrict__ ofc1w, const float* __restrict__ ofc2w,
                       const float* __restrict__ gwih,
                       h16* __restrict__ W1h, h16* __restrict__ W2h,
                       h16* __restrict__ O1h, h16* __restrict__ O2h,
                       h16* __restrict__ Gih) {
  int stride = gridDim.x * blockDim.x;
  int tid = blockIdx.x * blockDim.x + threadIdx.x;
  for (int i = tid; i < Kk * MHc * 32; i += stride) {           // (K,256,8)->(.,32)
    int kk = i & 31; int o = i >> 5;
    W1h[i] = (kk < 2 * Dd) ? (h16)fc1w[o * (2 * Dd) + kk] : (h16)0.f;
  }
  for (int i = tid; i < Kk * MOc * MHc; i += stride) W2h[i] = (h16)fc2w[i];
  for (int i = tid; i < NHc * H2p; i += stride) {               // (256,260)->(256,288)
    int c = i % H2p; int o = i / H2p;
    O1h[i] = (c < H2c) ? (h16)ofc1w[o * H2c + c] : (h16)0.f;
  }
  for (int i = tid; i < NHc * NHc; i += stride) O2h[i] = (h16)ofc2w[i];
  for (int i = tid; i < G3p * H2p; i += stride) {               // (780,260)->(784,288)
    int c = i % H2p; int o = i / H2p;
    Gih[i] = (o < G3 && c < H2c) ? (h16)gwih[o * H2c + c] : (h16)0.f;
  }
}

// ---------------- k_init: aug = [x | 0], hnh = 0 ---------------------------
__global__ void k_init(const float* __restrict__ inputs,
                       float* __restrict__ aug, h16* __restrict__ hnh) {
  size_t stride = (size_t)gridDim.x * blockDim.x;
  size_t tid = (size_t)blockIdx.x * blockDim.x + threadIdx.x;
  const size_t naug = (size_t)Bb * Tt * Nn * H2c;
  for (size_t i = tid; i < naug; i += stride) {
    int c = (int)(i % H2c); size_t r = i / H2c;
    int n = (int)(r % Nn); size_t r2 = r / Nn;
    int t = (int)(r2 % Tt); int b = (int)(r2 / Tt);
    aug[i] = (c < Dd) ? inputs[(((size_t)b * Nn + n) * Tt + t) * Dd + c] : 0.f;
  }
  const size_t nh = (size_t)Bb * Tt * Nn * H2p;
  for (size_t i = tid; i < nh; i += stride) hnh[i] = (h16)0.f;
}

// ---------------- k_prep_pre: pre = [x_send | x_recv] as f16 ---------------
__global__ void k_prep_pre(const float* __restrict__ inputs,
                           const int* __restrict__ send_idx, const int* __restrict__ recv_idx,
                           h16* __restrict__ preh) {
  size_t stride = (size_t)gridDim.x * blockDim.x;
  size_t tid = (size_t)blockIdx.x * blockDim.x + threadIdx.x;
  const size_t R = (size_t)Bb * Tt * Ee;
  for (size_t i = tid; i < R; i += stride) {
    int e = (int)(i % Ee); size_t r = i / Ee;
    int t = (int)(r % Tt); int b = (int)(r / Tt);
    int sn = send_idx[e], rn = recv_idx[e];
    const float* xs = inputs + (((size_t)b * Nn + sn) * Tt + t) * Dd;
    const float* xr = inputs + (((size_t)b * Nn + rn) * Tt + t) * Dd;
#pragma unroll
    for (int d = 0; d < Dd; ++d) {
      preh[i * 8 + d]      = (h16)xs[d];
      preh[i * 8 + Dd + d] = (h16)xr[d];
    }
  }
}

// ---------------- k_msg: fused edge MLP + scatter aggregation --------------
// One block (4 waves, 128 thr) per 16-edge tile of one (b,t). Each wave owns
// 4 of the 16 output column tiles -> accum[4] (32 VGPRs), no spills.
__global__ void k_msg(const h16* __restrict__ preh, const h16* __restrict__ W1h,
                      const h16* __restrict__ W2h, const float* __restrict__ fc1b,
                      const float* __restrict__ fc2b, const float* __restrict__ rel_type,
                      const int* __restrict__ recv_idx, float* __restrict__ aug) {
  __shared__ h16 m1[16][264];   // m1 tile staged for the fc2 A-fragment transpose
  const int tile = blockIdx.x;
  const int bt = tile / ETILES;             // b*T + t
  const int et = tile % ETILES;
  const int b  = bt / Tt;
  const int t  = bt % Tt;
  const int e0 = et * 16;
  const int lane = threadIdx.x & 31;
  const int wv   = threadIdx.x >> 5;        // 0..3: wave owns j = wv*4 .. wv*4+3
  const int l15  = lane & 15;
  const int half = lane >> 4;

  // A fragment: 16 edges x K=8 (zero-padded to 32)
  const h16* arow = preh + ((size_t)bt * Ee + e0 + l15) * 8;
  v16h apre = frag_ld_kmax(arow, half, 2 * Dd);

  v8f accum[4];
#pragma unroll
  for (int jj = 0; jj < 4; ++jj) accum[jj] = zero8();

  for (int k = 0; k < Kk; ++k) {
    // fc1: (16x8) @ (8x256) -> relu -> LDS (each wave writes its 4 tiles)
#pragma unroll
    for (int jj = 0; jj < 4; ++jj) {
      const int j = wv * 4 + jj;
      const h16* bp = W1h + ((size_t)(k * MHc + j * 16 + l15)) * 32;
      v8f c = wmma16(apre, frag_ld(bp, half), zero8());
      float bias = fc1b[k * MHc + j * 16 + l15];
#pragma unroll
      for (int r = 0; r < 8; ++r) {
        float v = c[r] + bias;
        m1[r + 8 * half][j * 16 + l15] = (h16)(v > 0.f ? v : 0.f);
      }
    }
    __syncthreads();

    // per-row rel_type weight for this k
    float rt[8];
#pragma unroll
    for (int r = 0; r < 8; ++r) {
      int e = e0 + r + 8 * half;
      rt[r] = rel_type[((size_t)b * Ee + e) * Kk + k];
    }

    // fc2: (16x256) @ (256x256), relu, weight, accumulate over k
#pragma unroll
    for (int jj = 0; jj < 4; ++jj) {
      const int j = wv * 4 + jj;
      v8f c = zero8();
#pragma unroll
      for (int ks = 0; ks < 8; ++ks) {
        v16h am = frag_ld(&m1[l15][ks * 32], half);
        const h16* bp = W2h + ((size_t)(k * MOc + j * 16 + l15)) * MHc + ks * 32;
        c = wmma16(am, frag_ld(bp, half), c);
      }
      float bias = fc2b[k * MOc + j * 16 + l15];
#pragma unroll
      for (int r = 0; r < 8; ++r) {
        float v = c[r] + bias;
        v = v > 0.f ? v : 0.f;
        accum[jj][r] += v * rt[r];
      }
    }
    __syncthreads();
  }

  // scatter-add into aug[b,t,recv(e), 4+col]  (agg einsum with one-hot rel_rec)
#pragma unroll
  for (int jj = 0; jj < 4; ++jj) {
    const int j = wv * 4 + jj;
#pragma unroll
    for (int r = 0; r < 8; ++r) {
      int e = e0 + r + 8 * half;
      int nrecv = recv_idx[e];
      float* dst = aug + (((size_t)b * Tt + t) * Nn + nrecv) * H2c + Dd + j * 16 + l15;
      atomicAdd(dst, accum[jj][r]);
    }
  }
}

// ---------------- k_mlp2: two tiny ELU layers on aug[:,0] ------------------
__global__ void k_mlp2(const float* __restrict__ aug,
                       const float* __restrict__ w1, const float* __restrict__ b1,
                       const float* __restrict__ w2, const float* __restrict__ b2,
                       float* __restrict__ hbuf) {
  __shared__ float row[H2c];
  __shared__ float mid[H2c];
  int idx = blockIdx.x;            // b*N + n
  int b = idx / Nn, n = idx % Nn;
  int tid = threadIdx.x;
  const float* ar = aug + (((size_t)b * Tt + 0) * Nn + n) * H2c;
  if (tid < H2c) row[tid] = ar[tid];
  __syncthreads();
  if (tid < H2c) {
    const float* wr = w1 + (size_t)tid * H2c;
    float s = b1[tid];
    for (int c = 0; c < H2c; ++c) s += wr[c] * row[c];
    mid[tid] = eluf(s);
  }
  __syncthreads();
  if (tid < H2c) {
    const float* wr = w2 + (size_t)tid * H2c;
    float s = b2[tid];
    for (int c = 0; c < H2c; ++c) s += wr[c] * mid[c];
    hbuf[(size_t)idx * H2c + tid] = eluf(s);
  }
}

// ---------------- k_bn: batchnorm over B; write h1 and hnh[:,0] ------------
__global__ void k_bn(const float* __restrict__ hbuf,
                     const float* __restrict__ g, const float* __restrict__ bta,
                     float* __restrict__ h1, h16* __restrict__ hnh) {
  int i = blockIdx.x * blockDim.x + threadIdx.x;   // (n,c)
  if (i >= Nn * H2c) return;
  int n = i / H2c, c = i % H2c;
  float mu = 0.f;
#pragma unroll
  for (int b = 0; b < Bb; ++b) mu += hbuf[((size_t)b * Nn + n) * H2c + c];
  mu *= (1.f / Bb);
  float var = 0.f;
#pragma unroll
  for (int b = 0; b < Bb; ++b) {
    float d = hbuf[((size_t)b * Nn + n) * H2c + c] - mu;
    var += d * d;
  }
  var *= (1.f / Bb);
  float inv = rsqrtf(var + BN_EPS);
#pragma unroll
  for (int b = 0; b < Bb; ++b) {
    float v = g[c] * (hbuf[((size_t)b * Nn + n) * H2c + c] - mu) * inv + bta[c];
    int idx = n * Bb + b;                 // h1 is (N,B,H)
    h1[(size_t)idx * H2c + c] = v;
    int b2 = idx >> 5, n2 = idx & 31;     // h1[None].reshape(B,1,N,H) mapping
    hnh[(((size_t)b2 * Tt + 0) * Nn + n2) * H2p + c] = (h16)v;
  }
}

// ---------------- k_augh: f16 padded copy of aug for WMMA A ----------------
__global__ void k_augh(const float* __restrict__ aug, h16* __restrict__ augh) {
  size_t stride = (size_t)gridDim.x * blockDim.x;
  size_t tid = (size_t)blockIdx.x * blockDim.x + threadIdx.x;
  const size_t nh = (size_t)Bb * Tt * Nn * H2p;
  for (size_t i = tid; i < nh; i += stride) {
    int c = (int)(i % H2p); size_t r = i / H2p;
    augh[i] = (c < H2c) ? (h16)aug[r * H2c + c] : (h16)0.f;
  }
}

// ---------------- k_gi: gi = xseq @ w_ih^T + b_ih (WMMA) -------------------
// rows = (T-1)*N*B = 12544, row = t*256 + n*8 + b; xt = aug[:, t+1]
__global__ void k_gi(const h16* __restrict__ augh, const h16* __restrict__ Gih,
                     const float* __restrict__ b_ih, float* __restrict__ gi) {
  const int lane = threadIdx.x;
  const int l15 = lane & 15, half = lane >> 4;
  const int r0 = blockIdx.x * 16;
  const int row = r0 + l15;
  const int t = row >> 8;             // 256 rows per step
  const int idx = row & 255;
  const int n = idx >> 3, b = idx & 7;
  const h16* ap = augh + (((size_t)b * Tt + (t + 1)) * Nn + n) * H2p;

  for (int j = 0; j < 49; ++j) {      // 784 output cols (780 valid)
    v8f c = zero8();
#pragma unroll
    for (int ks = 0; ks < 9; ++ks) {
      const h16* bp = Gih + ((size_t)(j * 16 + l15)) * H2p + ks * 32;
      c = wmma16(frag_ld(ap + ks * 32, half), frag_ld(bp, half), c);
    }
    int col = j * 16 + l15;
    if (col < G3) {
      float bias = b_ih[col];
#pragma unroll
      for (int r = 0; r < 8; ++r) {
        int m = r + 8 * half;
        gi[(size_t)(r0 + m) * G3 + col] = c[r] + bias;
      }
    }
  }
}

// ---------------- k_gru: sequential scan, one block per (n,b) row ----------
__global__ void k_gru(const float* __restrict__ h1, const float* __restrict__ gi,
                      const float* __restrict__ w_hh, const float* __restrict__ b_hh,
                      h16* __restrict__ hnh) {
  __shared__ float hprev[H2c];
  __shared__ float gh[G3];
  const int flat = blockIdx.x;            // n*B + b
  const int tid = threadIdx.x;
  if (tid < H2c) hprev[tid] = h1[(size_t)flat * H2c + tid];
  __syncthreads();

  for (int t = 0; t < Tt - 1; ++t) {
    for (int o = tid; o < G3; o += blockDim.x) {
      const float* wr = w_hh + (size_t)o * H2c;
      float s = b_hh[o];
      for (int c = 0; c < H2c; ++c) s += wr[c] * hprev[c];
      gh[o] = s;
    }
    __syncthreads();
    float hn_val = 0.f;
    if (tid < H2c) {
      const float* gr = gi + ((size_t)t * 256 + flat) * G3;
      float r = sigf(gr[tid] + gh[tid]);
      float z = sigf(gr[H2c + tid] + gh[H2c + tid]);
      float nn = tanhf(gr[2 * H2c + tid] + r * gh[2 * H2c + tid]);
      hn_val = (1.f - z) * nn + z * hprev[tid];
    }
    __syncthreads();
    if (tid < H2c) {
      hprev[tid] = hn_val;
      // hs.transpose(1,2,0,3).reshape(B,T-1,N,H) raw-reshape mapping
      int gidx = flat * (Tt - 1) + t;
      int n2 = gidx & 31;
      int q = gidx >> 5;
      int b2 = q / (Tt - 1);
      int t2 = q % (Tt - 1);
      hnh[(((size_t)b2 * Tt + 1 + t2) * Nn + n2) * H2p + tid] = (h16)hn_val;
    }
    __syncthreads();
  }
}

// ---------------- k_out: fused 3-layer output MLP + residual + transpose ---
__global__ void k_out(const h16* __restrict__ hnh, const h16* __restrict__ O1h,
                      const float* __restrict__ b1, const h16* __restrict__ O2h,
                      const float* __restrict__ b2, const float* __restrict__ w3,
                      const float* __restrict__ b3, const float* __restrict__ inputs,
                      float* __restrict__ out) {
  __shared__ h16 p1[16][272];
  __shared__ h16 p2[16][272];
  const int lane = threadIdx.x;
  const int l15 = lane & 15, half = lane >> 4;
  const int r0 = blockIdx.x * 16;           // rows = b*T*N flat, matches hnh layout
  const h16* ap = hnh + (size_t)(r0 + l15) * H2p;

  // layer 1: 260(->288) -> 256, relu
#pragma unroll
  for (int j = 0; j < 16; ++j) {
    v8f c = zero8();
#pragma unroll
    for (int ks = 0; ks < 9; ++ks) {
      const h16* bp = O1h + ((size_t)(j * 16 + l15)) * H2p + ks * 32;
      c = wmma16(frag_ld(ap + ks * 32, half), frag_ld(bp, half), c);
    }
    float bias = b1[j * 16 + l15];
#pragma unroll
    for (int r = 0; r < 8; ++r) {
      float v = c[r] + bias;
      p1[r + 8 * half][j * 16 + l15] = (h16)(v > 0.f ? v : 0.f);
    }
  }
  __syncthreads();

  // layer 2: 256 -> 256, relu
#pragma unroll
  for (int j = 0; j < 16; ++j) {
    v8f c = zero8();
#pragma unroll
    for (int ks = 0; ks < 8; ++ks) {
      const h16* bp = O2h + ((size_t)(j * 16 + l15)) * NHc + ks * 32;
      c = wmma16(frag_ld(&p1[l15][ks * 32], half), frag_ld(bp, half), c);
    }
    float bias = b2[j * 16 + l15];
#pragma unroll
    for (int r = 0; r < 8; ++r) {
      float v = c[r] + bias;
      p2[r + 8 * half][j * 16 + l15] = (h16)(v > 0.f ? v : 0.f);
    }
  }
  __syncthreads();

  // layer 3: 256 -> 4, residual, store transposed (B,N,T-1,D)
  for (int pr = lane; pr < 64; pr += 32) {
    int m = pr >> 2, d = pr & 3;
    const float* wr = w3 + (size_t)d * NHc;
    float s = b3[d];
    for (int c = 0; c < NHc; ++c) s += (float)p2[m][c] * wr[c];
    int row = r0 + m;
    int b = row / (Tt * Nn);
    int rr = row % (Tt * Nn);
    int t = rr / Nn, n = rr % Nn;
    if (t < Tt - 1) {
      size_t xo = (((size_t)b * Nn + n) * Tt + t) * Dd + d;
      out[(((size_t)b * Nn + n) * (Tt - 1) + t) * Dd + d] = inputs[xo] + s;
    }
  }
}

// ---------------------------------------------------------------------------
extern "C" void kernel_launch(void* const* d_in, const int* in_sizes, int n_in,
                              void* d_out, int out_size, void* d_ws, size_t ws_size,
                              hipStream_t stream) {
  (void)in_sizes; (void)n_in; (void)out_size; (void)ws_size;
  const float* inputs    = (const float*)d_in[0];
  const float* rel_type  = (const float*)d_in[1];
  const float* rel_rec   = (const float*)d_in[2];
  const float* rel_send  = (const float*)d_in[3];
  const float* msg_fc1_w = (const float*)d_in[4];
  const float* msg_fc1_b = (const float*)d_in[5];
  const float* msg_fc2_w = (const float*)d_in[6];
  const float* msg_fc2_b = (const float*)d_in[7];
  const float* mlp2_fc1_w = (const float*)d_in[8];
  const float* mlp2_fc1_b = (const float*)d_in[9];
  const float* mlp2_fc2_w = (const float*)d_in[10];
  const float* mlp2_fc2_b = (const float*)d_in[11];
  const float* mlp2_bn_g  = (const float*)d_in[12];
  const float* mlp2_bn_b  = (const float*)d_in[13];
  const float* gru_w_ih   = (const float*)d_in[14];
  const float* gru_w_hh   = (const float*)d_in[15];
  const float* gru_b_ih   = (const float*)d_in[16];
  const float* gru_b_hh   = (const float*)d_in[17];
  const float* out_fc1_w  = (const float*)d_in[18];
  const float* out_fc1_b  = (const float*)d_in[19];
  const float* out_fc2_w  = (const float*)d_in[20];
  const float* out_fc2_b  = (const float*)d_in[21];
  const float* out_fc3_w  = (const float*)d_in[22];
  const float* out_fc3_b  = (const float*)d_in[23];
  float* outp = (float*)d_out;

  // workspace carve (aligned to 256B)
  char* p = (char*)d_ws;
  auto take = [&](size_t bytes) -> char* {
    char* r = p;
    p += (bytes + 255) & ~(size_t)255;
    return r;
  };
  int* send_idx = (int*)take((size_t)Ee * 4);
  int* recv_idx = (int*)take((size_t)Ee * 4);
  h16* preh = (h16*)take((size_t)Bb * Tt * Ee * 8 * 2);
  h16* W1h  = (h16*)take((size_t)Kk * MHc * 32 * 2);
  h16* W2h  = (h16*)take((size_t)Kk * MOc * MHc * 2);
  h16* O1h  = (h16*)take((size_t)NHc * H2p * 2);
  h16* O2h  = (h16*)take((size_t)NHc * NHc * 2);
  h16* Gih  = (h16*)take((size_t)G3p * H2p * 2);
  float* aug  = (float*)take((size_t)Bb * Tt * Nn * H2c * 4);
  h16* augh   = (h16*)take((size_t)Bb * Tt * Nn * H2p * 2);
  float* hbuf = (float*)take((size_t)Bb * Nn * H2c * 4);
  float* h1   = (float*)take((size_t)Nn * Bb * H2c * 4);
  float* gi   = (float*)take((size_t)(Tt - 1) * Nn * Bb * G3 * 4);
  h16* hnh    = (h16*)take((size_t)Bb * Tt * Nn * H2p * 2);

  k_indices<<<(Ee + 255) / 256, 256, 0, stream>>>(rel_send, rel_rec, send_idx, recv_idx);
  k_pack<<<256, 256, 0, stream>>>(msg_fc1_w, msg_fc2_w, out_fc1_w, out_fc2_w, gru_w_ih,
                                  W1h, W2h, O1h, O2h, Gih);
  k_init<<<512, 256, 0, stream>>>(inputs, aug, hnh);
  k_prep_pre<<<1024, 256, 0, stream>>>(inputs, send_idx, recv_idx, preh);
  k_msg<<<Bb * Tt * ETILES, 128, 0, stream>>>(preh, W1h, W2h, msg_fc1_b, msg_fc2_b,
                                              rel_type, recv_idx, aug);
  k_mlp2<<<Bb * Nn, 288, 0, stream>>>(aug, mlp2_fc1_w, mlp2_fc1_b, mlp2_fc2_w,
                                      mlp2_fc2_b, hbuf);
  k_bn<<<(Nn * H2c + 255) / 256, 256, 0, stream>>>(hbuf, mlp2_bn_g, mlp2_bn_b, h1, hnh);
  k_augh<<<512, 256, 0, stream>>>(aug, augh);
  k_gi<<<(Tt - 1) * Nn * Bb / 16, 32, 0, stream>>>(augh, Gih, gru_b_ih, gi);
  k_gru<<<Nn * Bb, 288, 0, stream>>>(h1, gi, gru_w_hh, gru_b_hh, hnh);
  k_out<<<Bb * Tt * Nn / 16, 32, 0, stream>>>(hnh, O1h, out_fc1_b, O2h, out_fc2_b,
                                              out_fc3_w, out_fc3_b, inputs, outp);
}
